// MoELayer_8813272891795
// MI455X (gfx1250) — matrix operand order repaired and checked by reading, not compiled, not run on hardware
//
#include <hip/hip_runtime.h>

#define NTOK   32768
#define DIN    1024
#define DOUT   1024
#define NEXP   8

#define BM 128
#define BN 128
#define BK 32
#define LDK (BK + 8)           // padded LDS row stride (halfs), 80B = 16B-multiple
#define KSTEPS (DIN / BK)      // 32
#define CAP (NTOK * 2 + NEXP * BM)   // 66560 padded rows
#define MAXTILES (CAP / BM)          // 520

typedef __attribute__((ext_vector_type(16))) _Float16 halfx16;
typedef __attribute__((ext_vector_type(2)))  __fp16   fp16x2;   // cvt_pkrtz return type
typedef __attribute__((ext_vector_type(8)))  float  f32x8;
typedef __attribute__((ext_vector_type(4)))  float  f32x4;
typedef __attribute__((ext_vector_type(4)))  unsigned int u32x4;

union FragU { u32x4 q[2]; halfx16 v; };

// ---- init: zero output, counters; poison token slots with -1 --------------
__global__ void k_init(float* __restrict__ out, int* __restrict__ cnts16,
                       int* __restrict__ tok) {
  size_t i = (size_t)blockIdx.x * blockDim.x + threadIdx.x;
  size_t stride = (size_t)gridDim.x * blockDim.x;
  for (size_t j = i; j < (size_t)NTOK * DOUT; j += stride) out[j] = 0.0f;
  for (size_t j = i; j < (size_t)CAP; j += stride) tok[j] = -1;
  if (i < 16) cnts16[i] = 0;   // counts[8] + cursor[8]
}

// ---- gating: one wave per token ------------------------------------------
__global__ __launch_bounds__(256) void k_gate(
    const float* __restrict__ x, const float* __restrict__ Wg,
    const float* __restrict__ bg, int* __restrict__ pe,
    float* __restrict__ pw, int* __restrict__ counts) {
  const int wave = threadIdx.x >> 5, lane = threadIdx.x & 31;
  const int t = blockIdx.x * 8 + wave;
  const f32x4* x4 = (const f32x4*)(x + (size_t)t * DIN);
  float s[NEXP];
  #pragma unroll
  for (int e = 0; e < NEXP; ++e) s[e] = 0.0f;
  #pragma unroll
  for (int i = 0; i < 8; ++i) {
    f32x4 xv = x4[i * 32 + lane];
    #pragma unroll
    for (int e = 0; e < NEXP; ++e) {
      f32x4 wv = ((const f32x4*)(Wg + (size_t)e * DIN))[i * 32 + lane];
      s[e] += xv.x * wv.x + xv.y * wv.y + xv.z * wv.z + xv.w * wv.w;
    }
  }
  #pragma unroll
  for (int e = 0; e < NEXP; ++e) {
    #pragma unroll
    for (int off = 16; off > 0; off >>= 1) s[e] += __shfl_xor(s[e], off, 32);
  }
  if (lane == 0) {
    float m = -3.4e38f;
    #pragma unroll
    for (int e = 0; e < NEXP; ++e) { s[e] += bg[e]; m = fmaxf(m, s[e]); }
    float p[NEXP], sum = 0.0f;
    #pragma unroll
    for (int e = 0; e < NEXP; ++e) { p[e] = expf(s[e] - m); sum += p[e]; }
    float inv = 1.0f / sum;
    int e1 = 0; float b1 = p[0];
    #pragma unroll
    for (int e = 1; e < NEXP; ++e) if (p[e] > b1) { b1 = p[e]; e1 = e; }
    int e2 = -1; float b2 = -1.0f;
    #pragma unroll
    for (int e = 0; e < NEXP; ++e)
      if (e != e1 && p[e] > b2) { b2 = p[e]; e2 = e; }
    pe[2 * t] = e1;     pw[2 * t] = b1 * inv;
    pe[2 * t + 1] = e2; pw[2 * t + 1] = b2 * inv;
    atomicAdd(&counts[e1], 1);
    atomicAdd(&counts[e2], 1);
  }
}

// ---- scan: BM-aligned exclusive offsets per expert -----------------------
__global__ void k_scan(const int* __restrict__ counts, int* __restrict__ poff) {
  if (threadIdx.x == 0) {
    int acc = 0;
    #pragma unroll
    for (int e = 0; e < NEXP; ++e) {
      poff[e] = acc;
      acc += (counts[e] + BM - 1) & ~(BM - 1);
    }
    poff[NEXP] = acc;
  }
}

// ---- scatter tokens into expert-contiguous padded segments ---------------
__global__ __launch_bounds__(256) void k_scatter(
    const int* __restrict__ pe, const float* __restrict__ pw,
    const int* __restrict__ poff, int* __restrict__ cursor,
    int* __restrict__ tok, float* __restrict__ rw) {
  int t = blockIdx.x * 256 + threadIdx.x;
  #pragma unroll
  for (int j = 0; j < 2; ++j) {
    int e = pe[2 * t + j];
    int pos = poff[e] + atomicAdd(&cursor[e], 1);
    tok[pos] = t;
    rw[pos] = pw[2 * t + j];
  }
}

// ---- LDS staging helpers --------------------------------------------------
__device__ __forceinline__ void load16(const float* __restrict__ p, float* f) {
  #pragma unroll
  for (int j = 0; j < 4; ++j) {
    f32x4 v = *(const f32x4*)(p + 4 * j);
    f[4 * j + 0] = v.x; f[4 * j + 1] = v.y;
    f[4 * j + 2] = v.z; f[4 * j + 3] = v.w;
  }
}
// split 16 floats into f16 hi/lo pairs via v_cvt_pk_f16_f32; 32B to each array
__device__ __forceinline__ void store_hilo(unsigned short* dh, unsigned short* dl,
                                           const float* f) {
  u32x4 qh[2], ql[2];
  unsigned int* ph = (unsigned int*)qh;
  unsigned int* pl = (unsigned int*)ql;
  #pragma unroll
  for (int j = 0; j < 8; ++j) {
    float a = f[2 * j], b = f[2 * j + 1];
    fp16x2 hi = __builtin_amdgcn_cvt_pkrtz(a, b);          // v_cvt_pk_f16_f32
    float ra = a - (float)hi.x;                            // exact residual
    float rb = b - (float)hi.y;
    fp16x2 lo = __builtin_amdgcn_cvt_pkrtz(ra, rb);
    ph[j] = __builtin_bit_cast(unsigned int, hi);
    pl[j] = __builtin_bit_cast(unsigned int, lo);
  }
  ((u32x4*)dh)[0] = qh[0]; ((u32x4*)dh)[1] = qh[1];
  ((u32x4*)dl)[0] = ql[0]; ((u32x4*)dl)[1] = ql[1];
}
// A frag (16x32 f16): lanes0-15 K0..7 & K16..23, lanes16-31 K8..15 & K24..31
__device__ __forceinline__ halfx16 fragA(const unsigned short* base, int lhi) {
  FragU f;
  f.q[0] = *(const u32x4*)(base + lhi * 8);
  f.q[1] = *(const u32x4*)(base + 16 + lhi * 8);
  return f.v;
}
// B frag (32x16 f16): lanes0-15 K0..7 & K16..23, lanes16-31 K8..15 & K24..31
__device__ __forceinline__ halfx16 fragB(const unsigned short* base, int lhi) {
  FragU f;
  f.q[0] = *(const u32x4*)(base + lhi * 8);
  f.q[1] = *(const u32x4*)(base + 16 + lhi * 8);
  return f.v;
}

// ---- grouped GEMM: out[tok] += w * (x_row @ W[e]^T + b[e]) ---------------
__global__ __launch_bounds__(256) void k_gemm(
    const float* __restrict__ x, const float* __restrict__ W,
    const float* __restrict__ bvec, const int* __restrict__ tok,
    const float* __restrict__ rw, const int* __restrict__ poff,
    float* __restrict__ out) {
  __shared__ __align__(16) unsigned short sAh[BM][LDK];
  __shared__ __align__(16) unsigned short sAl[BM][LDK];
  __shared__ __align__(16) unsigned short sBh[BN][LDK];
  __shared__ __align__(16) unsigned short sBl[BN][LDK];
  __shared__ int   tok_s[BM];
  __shared__ float w_s[BM];

  const int rowbase = blockIdx.x * BM;
  if (rowbase >= poff[NEXP]) return;
  int e = 0;
  #pragma unroll
  for (int i = 0; i < NEXP; ++i) if (rowbase >= poff[i]) e = i;
  const int nbase = blockIdx.y * BN;

  const int t = threadIdx.x;
  if (t < BM) {
    tok_s[t] = tok[rowbase + t];
    w_s[t] = rw[rowbase + t];
  }
  __syncthreads();

  // staging: 2 threads per row, each covers 16 contiguous f32 of the k-slab
  const int srow = t >> 1, shalf = t & 1;
  int stok = tok_s[srow]; if (stok < 0) stok = 0;  // padding rows read safe data
  const float* xp = x + (size_t)stok * DIN + shalf * 16;
  const float* wp = W + ((size_t)e * DOUT + nbase + srow) * DIN + shalf * 16;
  unsigned short* dAh = &sAh[srow][shalf * 16];
  unsigned short* dAl = &sAl[srow][shalf * 16];
  unsigned short* dBh = &sBh[srow][shalf * 16];
  unsigned short* dBl = &sBl[srow][shalf * 16];

  const int wave = t >> 5, lane = t & 31;
  const int wm = wave & 3, wn = wave >> 2;     // 4x2 wave grid: 32x64 per wave
  const int l16 = lane & 15, lhi = lane >> 4;
  const unsigned short* pAh = &sAh[wm * 32 + l16][0];
  const unsigned short* pAl = &sAl[wm * 32 + l16][0];
  const unsigned short* pBh = &sBh[wn * 64 + l16][0];
  const unsigned short* pBl = &sBl[wn * 64 + l16][0];

  f32x8 acc[2][4];
  const f32x8 zero = {0.f, 0.f, 0.f, 0.f, 0.f, 0.f, 0.f, 0.f};
  #pragma unroll
  for (int mi = 0; mi < 2; ++mi)
    #pragma unroll
    for (int ni = 0; ni < 4; ++ni) acc[mi][ni] = zero;

  float fa[16], fb[16];
  load16(xp, fa);
  load16(wp, fb);

  for (int kt = 0; kt < KSTEPS; ++kt) {
    __syncthreads();                 // previous tile fully consumed
    store_hilo(dAh, dAl, fa);
    store_hilo(dBh, dBl, fb);
    __syncthreads();                 // tile visible to all waves
    if (kt + 1 < KSTEPS) {           // overlap next global fetch with compute
      load16(xp + (kt + 1) * BK, fa);
      load16(wp + (kt + 1) * BK, fb);
      if (kt + 2 < KSTEPS) {
        __builtin_prefetch(xp + (kt + 2) * BK, 0, 0);
        __builtin_prefetch(wp + (kt + 2) * BK, 0, 0);
      }
    }
    halfx16 Ah[2], Al[2];
    #pragma unroll
    for (int mi = 0; mi < 2; ++mi) {
      Ah[mi] = fragA(pAh + mi * 16 * LDK, lhi);
      Al[mi] = fragA(pAl + mi * 16 * LDK, lhi);
    }
    #pragma unroll
    for (int ni = 0; ni < 4; ++ni) {
      halfx16 Bh = fragB(pBh + ni * 16 * LDK, lhi);
      halfx16 Bl = fragB(pBl + ni * 16 * LDK, lhi);
      #pragma unroll
      for (int mi = 0; mi < 2; ++mi) {
        // near-fp32 via 3-term f16 split: Al*Bh + Ah*Bl + Ah*Bh
        acc[mi][ni] = __builtin_amdgcn_wmma_f32_16x16x32_f16(
            false, Al[mi], false, Bh, (short)0, acc[mi][ni], false, false);
        acc[mi][ni] = __builtin_amdgcn_wmma_f32_16x16x32_f16(
            false, Ah[mi], false, Bl, (short)0, acc[mi][ni], false, false);
        acc[mi][ni] = __builtin_amdgcn_wmma_f32_16x16x32_f16(
            false, Ah[mi], false, Bh, (short)0, acc[mi][ni], false, false);
      }
    }
  }

  // epilogue: C/D layout -> lane holds col n=l16, rows r+8*lhi
  #pragma unroll
  for (int mi = 0; mi < 2; ++mi) {
    const int mrow0 = wm * 32 + mi * 16 + lhi * 8;
    int tv[8]; float wv[8];
    #pragma unroll
    for (int r = 0; r < 8; ++r) { tv[r] = tok_s[mrow0 + r]; wv[r] = w_s[mrow0 + r]; }
    #pragma unroll
    for (int ni = 0; ni < 4; ++ni) {
      const int col = nbase + wn * 64 + ni * 16 + l16;
      const float cb = bvec[(size_t)e * DOUT + col];
      #pragma unroll
      for (int r = 0; r < 8; ++r) {
        if (tv[r] >= 0) {
          float val = wv[r] * (acc[mi][ni][r] + cb);
          __hip_atomic_fetch_add(out + (size_t)tv[r] * DOUT + col, val,
                                 __ATOMIC_RELAXED, __HIP_MEMORY_SCOPE_AGENT);
        }
      }
    }
  }
}

extern "C" void kernel_launch(void* const* d_in, const int* in_sizes, int n_in,
                              void* d_out, int out_size, void* d_ws, size_t ws_size,
                              hipStream_t stream) {
  const float* x  = (const float*)d_in[0];
  const float* Wg = (const float*)d_in[1];
  const float* bg = (const float*)d_in[2];
  const float* W  = (const float*)d_in[3];
  const float* b  = (const float*)d_in[4];
  float* out = (float*)d_out;

  int* cnts   = (int*)d_ws;        // counts[8]
  int* cursor = cnts + 8;          // cursor[8]
  int* poff   = cnts + 16;         // poff[9]
  int* pe     = cnts + 32;         // 2*NTOK expert ids
  float* pw   = (float*)(pe + 2 * NTOK);
  int* tokb   = (int*)(pw + 2 * NTOK);
  float* rwb  = (float*)(tokb + CAP);

  k_init<<<2048, 256, 0, stream>>>(out, cnts, tokb);
  k_gate<<<NTOK / 8, 256, 0, stream>>>(x, Wg, bg, pe, pw, cnts);
  k_scan<<<1, 32, 0, stream>>>(cnts, poff);
  k_scatter<<<NTOK / 256, 256, 0, stream>>>(pe, pw, poff, cursor, tokb, rwb);
  dim3 g(MAXTILES, DOUT / BN);
  k_gemm<<<g, 256, 0, stream>>>(x, W, b, tokb, rwb, poff, out);
}
